// CNFROM_15461882265686
// MI455X (gfx1250) — compile-verified, matrix operand order checked by reading, not run on hardware
//
#include <hip/hip_runtime.h>
#include <hip/hip_bf16.h>

typedef __attribute__((ext_vector_type(16))) _Float16 v16h;
typedef __attribute__((ext_vector_type(8)))  _Float16 v8h;
typedef __attribute__((ext_vector_type(8)))  float    v8f;

#define LATENT 10
#define HIDDEN 128
#define STEPS  101
#define HSTEP  (1.0f/100.0f)
#define PI_F   3.14159265358979f

union V16H { v16h v; v8h h[2]; _Float16 e[16]; };
struct F2 { v8f f[2]; };

// tanh: prefer gfx1250 HW tanh (single TRANS op); else exp2-based (~6 VALU).
// No clamp needed: |preact| is bounded (~20) by the Glorot weights + tanh-bounded
// activations, far below the exp2 overflow threshold (|x| > 44).
__device__ __forceinline__ float fast_tanh(float x) {
#if __has_builtin(__builtin_amdgcn_tanhf)
    return __builtin_amdgcn_tanhf(x);
#elif __has_builtin(__builtin_amdgcn_tanh_f32)
    return __builtin_amdgcn_tanh_f32(x);
#else
    float e = __builtin_amdgcn_exp2f(x * -2.8853900817779268f); // exp(-2x)
    return (1.0f - e) * __builtin_amdgcn_rcpf(1.0f + e);
#endif
}

// ---------------- fragment loaders ----------------
__device__ __forceinline__ v16h ldB(const _Float16* base, int frag, int lane) {
    const _Float16* p = base + frag * 512 + lane * 16;
    V16H u;
    u.h[0] = *(const v8h*)p;
    u.h[1] = *(const v8h*)(p + 8);
    return u.v;
}
// A fragment from a row-major 16xHIDDEN f16 tile at z (row 0 of the M-tile)
__device__ __forceinline__ v16h ldA(const _Float16* z, int kc, int lane) {
    const int m  = lane & 15;
    const int kb = kc * 32 + ((lane >> 4) << 3);
    V16H u;
    u.h[0] = *(const v8h*)(z + m * HIDDEN + kb);
    u.h[1] = *(const v8h*)(z + m * HIDDEN + kb + 16);
    return u.v;
}

// ---------------- two-M-tile 32x128 @ K x128 layer ----------------
template<int KC>
__device__ __forceinline__ void layer128x2(_Float16* z, const _Float16* wf,
                                           const float* bias, int lane, bool act) {
    v16h A[2][KC];
#pragma unroll
    for (int mt = 0; mt < 2; ++mt)
#pragma unroll
        for (int kc = 0; kc < KC; ++kc)
            A[mt][kc] = ldA(z + mt * 16 * HIDDEN, kc, lane);
    const int n   = lane & 15;
    const int mhi = (lane >= 16) ? 8 : 0;
#pragma unroll
    for (int nt = 0; nt < 8; ++nt) {
        v8f c0 = {}, c1 = {};
#pragma unroll
        for (int kc = 0; kc < KC; ++kc) {
            v16h Bf = ldB(wf, kc * 8 + nt, lane);
            c0 = __builtin_amdgcn_wmma_f32_16x16x32_f16(false, A[0][kc], false, Bf,
                                                        (short)0, c0, false, false);
            c1 = __builtin_amdgcn_wmma_f32_16x16x32_f16(false, A[1][kc], false, Bf,
                                                        (short)0, c1, false, false);
        }
        float bv = bias[nt * 16 + n];
#pragma unroll
        for (int r = 0; r < 8; ++r) {
            float v0 = c0[r] + bv, v1 = c1[r] + bv;
            if (act) { v0 = fast_tanh(v0); v1 = fast_tanh(v1); }
            z[(r + mhi) * HIDDEN + nt * 16 + n]      = (_Float16)v0;
            z[(16 + r + mhi) * HIDDEN + nt * 16 + n] = (_Float16)v1;
        }
    }
}

// ---------------- pnode on 2 M-tiles ----------------
__device__ __forceinline__ F2 pnode2(const F2& ain, float tcur, const float* murs,
                                     _Float16* zb, _Float16* ab,
                                     const _Float16* w1, const _Float16* w2, const _Float16* w3,
                                     const float* b1, const float* b2, const float* b3,
                                     int lane) {
    const int n   = lane & 15;
    const int mhi = (lane >= 16) ? 8 : 0;
#pragma unroll
    for (int mt = 0; mt < 2; ++mt)
#pragma unroll
        for (int r = 0; r < 8; ++r)
            ab[(mt * 16 + r + mhi) * 16 + n] = (_Float16)ain.f[mt][r];

    // A0 fragments (16x32: [alpha(10), t, mu, 0 pad])
    V16H A0[2];
    const int m  = lane & 15;
    const int kh = (lane >> 4) * 8;
#pragma unroll
    for (int mt = 0; mt < 2; ++mt) {
#pragma unroll
        for (int e = 0; e < 16; ++e) {
            int k = kh + (e & 7) + ((e >= 8) ? 16 : 0);
            float v;
            if (k < LATENT)           v = (float)ab[(mt * 16 + m) * 16 + k];
            else if (k == LATENT)     v = tcur;
            else if (k == LATENT + 1) v = murs[mt];
            else                      v = 0.0f;
            A0[mt].e[e] = (_Float16)v;
        }
    }
    // L1: 12(pad32) -> 128
#pragma unroll
    for (int nt = 0; nt < 8; ++nt) {
        v16h Bf = ldB(w1, nt, lane);
        v8f c0 = {}, c1 = {};
        c0 = __builtin_amdgcn_wmma_f32_16x16x32_f16(false, A0[0].v, false, Bf,
                                                    (short)0, c0, false, false);
        c1 = __builtin_amdgcn_wmma_f32_16x16x32_f16(false, A0[1].v, false, Bf,
                                                    (short)0, c1, false, false);
        float bv = b1[nt * 16 + n];
#pragma unroll
        for (int r = 0; r < 8; ++r) {
            zb[(r + mhi) * HIDDEN + nt * 16 + n]      = (_Float16)fast_tanh(c0[r] + bv);
            zb[(16 + r + mhi) * HIDDEN + nt * 16 + n] = (_Float16)fast_tanh(c1[r] + bv);
        }
    }
    // L2: 128 -> 128
    layer128x2<4>(zb, w2, b2, lane, true);
    // L3: 128 -> 10 (pad 16), linear
    v8f c0 = {}, c1 = {};
#pragma unroll
    for (int kc = 0; kc < 4; ++kc) {
        v16h Bf = ldB(w3, kc, lane);
        v16h Aa = ldA(zb, kc, lane);
        v16h Ab = ldA(zb + 16 * HIDDEN, kc, lane);
        c0 = __builtin_amdgcn_wmma_f32_16x16x32_f16(false, Aa, false, Bf, (short)0, c0, false, false);
        c1 = __builtin_amdgcn_wmma_f32_16x16x32_f16(false, Ab, false, Bf, (short)0, c1, false, false);
    }
    float bv = b3[n];
    F2 out;
#pragma unroll
    for (int r = 0; r < 8; ++r) { out.f[0][r] = c0[r] + bv; out.f[1][r] = c1[r] + bv; }
    return out;
}

// ---------------- weight repack: f32 row-major -> f16 B-fragments ----------------
__global__ void pack_frag(const float* __restrict__ W, int K, int N, int nTiles,
                          _Float16* __restrict__ dst) {
    const int frag = blockIdx.x;
    const int kc   = frag / nTiles;
    const int nt   = frag % nTiles;
    const int lane = threadIdx.x;
    const int n    = nt * 16 + (lane & 15);
    const int kh   = (lane >> 4) * 8;
    _Float16* p = dst + frag * 512 + lane * 16;
#pragma unroll
    for (int e = 0; e < 16; ++e) {
        int k = kc * 32 + kh + (e & 7) + ((e >= 8) ? 16 : 0);
        float v = (k < K && n < N) ? W[k * N + n] : 0.0f;
        p[e] = (_Float16)v;
    }
}

// ---------------- main kernel: 2 waves/block, 32 samples/wave ----------------
__global__ __launch_bounds__(64) void cnf_main(
    const float* __restrict__ x, const float* __restrict__ t, const float* __restrict__ mu,
    const float* __restrict__ pb1, const float* __restrict__ pb2, const float* __restrict__ pb3,
    const float* __restrict__ db1, const float* __restrict__ db2, const float* __restrict__ db3,
    const float* __restrict__ dW4, const float* __restrict__ db4,
    const _Float16* __restrict__ pw1f, const _Float16* __restrict__ pw2f,
    const _Float16* __restrict__ pw3f, const _Float16* __restrict__ dw1f,
    const _Float16* __restrict__ dw2f, const _Float16* __restrict__ dw3f,
    float* __restrict__ out)
{
    __shared__ alignas(32) _Float16 sW1[8 * 512];         //  8 KB
    __shared__ alignas(32) _Float16 sW2[32 * 512];        // 32 KB
    __shared__ alignas(32) _Float16 sW3[4 * 512];         //  4 KB
    __shared__ float sb1[128], sb2[128], sb3[16];
    __shared__ alignas(32) _Float16 zbuf[2][32 * HIDDEN]; // 16 KB (2 waves x 32 rows)
    __shared__ alignas(32) _Float16 abuf[2][32 * 16];     //  2 KB

    const int tid = threadIdx.x;
    for (int i = tid; i < 8 * 512;  i += 64) sW1[i] = pw1f[i];
    for (int i = tid; i < 32 * 512; i += 64) sW2[i] = pw2f[i];
    for (int i = tid; i < 4 * 512;  i += 64) sW3[i] = pw3f[i];
    for (int i = tid; i < 128; i += 64) { sb1[i] = pb1[i]; sb2[i] = pb2[i]; }
    if (tid < 16) sb3[tid] = (tid < LATENT) ? pb3[tid] : 0.0f;
    __syncthreads();

    const int lane = tid & 31;
    const int w    = tid >> 5;
    const int s0   = blockIdx.x * 64 + w * 32;   // 512 blocks * 2 waves * 32 samples
    _Float16* zb = &zbuf[w][0];
    _Float16* ab = &abuf[w][0];

    const int n   = lane & 15;
    const int mhi = (lane >= 16) ? 8 : 0;

    float murs[2], xrs[2], trs[2];
#pragma unroll
    for (int mt = 0; mt < 2; ++mt) {
        murs[mt] = mu[s0 + mt * 16 + n];
        xrs[mt]  = x[s0 + mt * 16 + n];
        trs[mt]  = t[s0 + mt * 16 + n];
    }

    int   idxr[2][8];
    float ratr[2][8];
#pragma unroll
    for (int mt = 0; mt < 2; ++mt)
#pragma unroll
        for (int r = 0; r < 8; ++r) {
            float tm = t[s0 + mt * 16 + r + mhi];
            float s  = tm * (float)(STEPS - 1);
            int id = (int)floorf(s);
            id = min(max(id, 0), STEPS - 2);
            idxr[mt][r] = id;
            ratr[mt][r] = s - (float)id;
        }

    // RK4 integration, capture traj[idx] / traj[idx+1] on the fly
    F2 alpha, asF, aeF;
#pragma unroll
    for (int mt = 0; mt < 2; ++mt) { alpha.f[mt] = v8f{}; asF.f[mt] = v8f{}; aeF.f[mt] = v8f{}; }

    for (int s = 0; s < STEPS; ++s) {
#pragma unroll
        for (int mt = 0; mt < 2; ++mt)
#pragma unroll
            for (int r = 0; r < 8; ++r) {
                if (idxr[mt][r] == s)     asF.f[mt][r] = alpha.f[mt][r];
                if (idxr[mt][r] + 1 == s) aeF.f[mt][r] = alpha.f[mt][r];
            }
        if (s == STEPS - 1) break;
        const float t0 = s * HSTEP;
        F2 k1 = pnode2(alpha, t0, murs, zb, ab, sW1, sW2, sW3, sb1, sb2, sb3, lane);
        F2 tmp;
#pragma unroll
        for (int mt = 0; mt < 2; ++mt)
#pragma unroll
            for (int r = 0; r < 8; ++r) tmp.f[mt][r] = alpha.f[mt][r] + (HSTEP / 3.0f) * k1.f[mt][r];
        F2 k2 = pnode2(tmp, t0 + HSTEP / 3.0f, murs, zb, ab, sW1, sW2, sW3, sb1, sb2, sb3, lane);
#pragma unroll
        for (int mt = 0; mt < 2; ++mt)
#pragma unroll
            for (int r = 0; r < 8; ++r) tmp.f[mt][r] = alpha.f[mt][r] + HSTEP * (k2.f[mt][r] - k1.f[mt][r] / 3.0f);
        F2 k3 = pnode2(tmp, t0 + 2.0f * HSTEP / 3.0f, murs, zb, ab, sW1, sW2, sW3, sb1, sb2, sb3, lane);
#pragma unroll
        for (int mt = 0; mt < 2; ++mt)
#pragma unroll
            for (int r = 0; r < 8; ++r) tmp.f[mt][r] = alpha.f[mt][r] + HSTEP * (k1.f[mt][r] - k2.f[mt][r] + k3.f[mt][r]);
        F2 k4 = pnode2(tmp, t0 + HSTEP, murs, zb, ab, sW1, sW2, sW3, sb1, sb2, sb3, lane);
#pragma unroll
        for (int mt = 0; mt < 2; ++mt)
#pragma unroll
            for (int r = 0; r < 8; ++r)
                alpha.f[mt][r] += HSTEP * (k1.f[mt][r] + 3.0f * k2.f[mt][r] + 3.0f * k3.f[mt][r] + k4.f[mt][r]) / 8.0f;
    }

    F2 ai;
#pragma unroll
    for (int mt = 0; mt < 2; ++mt)
#pragma unroll
        for (int r = 0; r < 8; ++r)
            ai.f[mt][r] = asF.f[mt][r] + ratr[mt][r] * (aeF.f[mt][r] - asF.f[mt][r]);

    // decoder input: [sin(16) | cos(16) | alpha(10)+pad | pad] per row (stride HIDDEN)
    {
        const int fb = (lane >> 4) * 8;
#pragma unroll
        for (int mt = 0; mt < 2; ++mt) {
            _Float16* zr = zb + (mt * 16 + n) * HIDDEN;
#pragma unroll
            for (int j = 0; j < 8; ++j) {
                float f   = 1.0f + 9.0f * (float)(fb + j) / 15.0f;   // linspace(1,10,16)
                float ang = 2.0f * PI_F * xrs[mt] * f;
                zr[fb + j]      = (_Float16)sinf(ang);
                zr[16 + fb + j] = (_Float16)cosf(ang);
                zr[48 + fb + j] = (_Float16)0.0f;
            }
#pragma unroll
            for (int r = 0; r < 8; ++r)
                zb[(mt * 16 + r + mhi) * HIDDEN + 32 + n] = (_Float16)ai.f[mt][r];
        }
    }

    layer128x2<2>(zb, dw1f, db1, lane, true);   // 42(pad64) -> 128
    layer128x2<4>(zb, dw2f, db2, lane, true);   // 128 -> 128
    layer128x2<4>(zb, dw3f, db3, lane, true);   // 128 -> 128

    // 128 -> 1 projection + output
    {
        const int half = (lane >> 4);
#pragma unroll
        for (int mt = 0; mt < 2; ++mt) {
            const _Float16* zr = zb + (mt * 16 + n) * HIDDEN;
            float p = 0.0f;
#pragma unroll
            for (int k = 0; k < 64; ++k)
                p += (float)zr[half * 64 + k] * dW4[half * 64 + k];
            float tot = p + __shfl_xor(p, 16, 32);
            if (lane < 16)
                out[s0 + mt * 16 + n] = -sinf(PI_F * xrs[mt]) + trs[mt] * (tot + db4[0]);
        }
    }
}

extern "C" void kernel_launch(void* const* d_in, const int* in_sizes, int n_in,
                              void* d_out, int out_size, void* d_ws, size_t ws_size,
                              hipStream_t stream) {
    (void)in_sizes; (void)n_in; (void)out_size; (void)ws_size;
    const float* x   = (const float*)d_in[0];
    const float* t   = (const float*)d_in[1];
    const float* mu  = (const float*)d_in[2];
    const float* pW1 = (const float*)d_in[3];  const float* pb1 = (const float*)d_in[4];
    const float* pW2 = (const float*)d_in[5];  const float* pb2 = (const float*)d_in[6];
    const float* pW3 = (const float*)d_in[7];  const float* pb3 = (const float*)d_in[8];
    const float* dW1 = (const float*)d_in[9];  const float* db1 = (const float*)d_in[10];
    const float* dW2 = (const float*)d_in[11]; const float* db2 = (const float*)d_in[12];
    const float* dW3 = (const float*)d_in[13]; const float* db3 = (const float*)d_in[14];
    const float* dW4 = (const float*)d_in[15]; const float* db4 = (const float*)d_in[16];

    char* ws = (char*)d_ws;
    _Float16* pw1f = (_Float16*)(ws + 0);        //  8 frags  =  8 KB
    _Float16* pw2f = (_Float16*)(ws + 8192);     // 32 frags  = 32 KB
    _Float16* pw3f = (_Float16*)(ws + 40960);    //  4 frags  =  4 KB
    _Float16* dw1f = (_Float16*)(ws + 45056);    // 16 frags  = 16 KB
    _Float16* dw2f = (_Float16*)(ws + 61440);    // 32 frags  = 32 KB
    _Float16* dw3f = (_Float16*)(ws + 94208);    // 32 frags  = 32 KB

    pack_frag<<<8,  32, 0, stream>>>(pW1, 12,  128, 8, pw1f);
    pack_frag<<<32, 32, 0, stream>>>(pW2, 128, 128, 8, pw2f);
    pack_frag<<<4,  32, 0, stream>>>(pW3, 128, 10,  1, pw3f);
    pack_frag<<<16, 32, 0, stream>>>(dW1, 42,  128, 8, dw1f);
    pack_frag<<<32, 32, 0, stream>>>(dW2, 128, 128, 8, dw2f);
    pack_frag<<<32, 32, 0, stream>>>(dW3, 128, 128, 8, dw3f);

    cnf_main<<<512, 64, 0, stream>>>(x, t, mu, pb1, pb2, pb3, db1, db2, db3,
                                     dW4, db4, pw1f, pw2f, pw3f, dw1f, dw2f, dw3f,
                                     (float*)d_out);
}
